// BinClassDecoder_73495480369364
// MI455X (gfx1250) — matrix operand order repaired and checked by reading, not compiled, not run on hardware
//
#include <hip/hip_runtime.h>
#include <math.h>

// ---------------------------------------------------------------------------
// MI455X (gfx1250). All K=512 contractions -> v_wmma_f32_16x16x32_f16 with
// f32 accumulate. Operands pre-staged as f16 (L2-resident) so the GEMM inner
// loop is pure b128 f16 loads + 4 WMMAs per 32-k step (16x64 strip per wave).
// The k-loop is software-pipelined: next iteration's fragments are preloaded
// into distinct registers before the current WMMA group, so VMEM latency is
// hidden under XDL work instead of a full s_wait_loadcnt 0 drain per WMMA.
// Additive-attention tanh score stage stays VALU (wave32 shuffle reductions).
// ---------------------------------------------------------------------------

typedef __attribute__((ext_vector_type(16))) _Float16 v16h;
typedef __attribute__((ext_vector_type(8)))  _Float16 v8h;
typedef __attribute__((ext_vector_type(8)))  float    v8f;

constexpr int Bc = 8, Sc = 512, Tc = 64, ENCc = 512, WORDc = 512;

// A fragment (16x32 f16): lane row = m0+l15; element j -> k = j + 16*half
__device__ __forceinline__ v16h loadA(const _Float16* __restrict__ arow,
                                      int k0, int half) {
  return *(const v16h*)(arow + k0 + 16 * half);
}

// ---------------------------------------------------------------------------
// Batched  C[M,N] = A[M,K] @ W[N,K]^T (+ bias[N]),  all-f16 operands, f32 out.
// One wave per 16(M) x 64(N) strip: A fragment reused across 4 B fragments.
// blockIdx.y selects batch via element strides (0 strides => non-batched).
// ---------------------------------------------------------------------------
__global__ __launch_bounds__(128) void gemm_f16(
    const _Float16* __restrict__ A, const _Float16* __restrict__ W,
    const float* __restrict__ bias, float* __restrict__ C,
    int M, int N, int K, long sA, long sW, long sC) {
  A += (size_t)blockIdx.y * sA;
  W += (size_t)blockIdx.y * sW;
  C += (size_t)blockIdx.y * sC;

  const int wavesPerBlock = blockDim.x >> 5;
  const int waveId = blockIdx.x * wavesPerBlock + (threadIdx.x >> 5);
  const int lane = threadIdx.x & 31;
  const int tilesN = N >> 6;                       // 64-wide strips
  const int tileM = waveId / tilesN;
  const int tileN = waveId % tilesN;
  if (tileM >= (M >> 4)) return;                   // wave-uniform guard
  const int m0 = tileM << 4, n0 = tileN << 6;
  const int half = lane >> 4, l15 = lane & 15;

  const _Float16* __restrict__ arow = A + (size_t)(m0 + l15) * K;
  // B fragment i: col n = n0 + 16*i + l15; element j -> k = j + 16*half
  const _Float16* __restrict__ w0 = W + (size_t)(n0 + 0  + l15) * K + 16 * half;
  const _Float16* __restrict__ w1 = W + (size_t)(n0 + 16 + l15) * K + 16 * half;
  const _Float16* __restrict__ w2 = W + (size_t)(n0 + 32 + l15) * K + 16 * half;
  const _Float16* __restrict__ w3 = W + (size_t)(n0 + 48 + l15) * K + 16 * half;

  // Prologue: fragments for k0 = 0
  v16h a  = loadA(arow, 0, half);
  v16h b0 = *(const v16h*)(w0);
  v16h b1 = *(const v16h*)(w1);
  v16h b2 = *(const v16h*)(w2);
  v16h b3 = *(const v16h*)(w3);

  v8f acc0 = {}, acc1 = {}, acc2 = {}, acc3 = {};
  // Software-pipelined steady state: preload k0+32 while WMMA'ing k0.
  for (int k0 = 0; k0 + 32 < K; k0 += 32) {
    const int kn = k0 + 32;
    __builtin_prefetch(arow + kn + 96, 0, 3);      // global_prefetch_b8
    __builtin_prefetch(w0 + kn + 96, 0, 3);
    v16h an = loadA(arow, kn, half);
    v16h c0 = *(const v16h*)(w0 + kn);
    v16h c1 = *(const v16h*)(w1 + kn);
    v16h c2 = *(const v16h*)(w2 + kn);
    v16h c3 = *(const v16h*)(w3 + kn);
    acc0 = __builtin_amdgcn_wmma_f32_16x16x32_f16(false, a, false, b0, (short)0, acc0, false, false);
    acc1 = __builtin_amdgcn_wmma_f32_16x16x32_f16(false, a, false, b1, (short)0, acc1, false, false);
    acc2 = __builtin_amdgcn_wmma_f32_16x16x32_f16(false, a, false, b2, (short)0, acc2, false, false);
    acc3 = __builtin_amdgcn_wmma_f32_16x16x32_f16(false, a, false, b3, (short)0, acc3, false, false);
    a = an; b0 = c0; b1 = c1; b2 = c2; b3 = c3;
  }
  // Epilogue k-step
  acc0 = __builtin_amdgcn_wmma_f32_16x16x32_f16(false, a, false, b0, (short)0, acc0, false, false);
  acc1 = __builtin_amdgcn_wmma_f32_16x16x32_f16(false, a, false, b1, (short)0, acc1, false, false);
  acc2 = __builtin_amdgcn_wmma_f32_16x16x32_f16(false, a, false, b2, (short)0, acc2, false, false);
  acc3 = __builtin_amdgcn_wmma_f32_16x16x32_f16(false, a, false, b3, (short)0, acc3, false, false);

  // C/D layout: VGPR r -> row m0 + r + 8*half, col = strip col + l15
  v8f accs[4] = {acc0, acc1, acc2, acc3};
#pragma unroll
  for (int i = 0; i < 4; ++i) {
    const int n = n0 + 16 * i + l15;
    const float bb = bias ? bias[n] : 0.f;
#pragma unroll
    for (int r = 0; r < 8; ++r)
      C[(size_t)(m0 + r + 8 * half) * N + n] = accs[i][r] + bb;
  }
}

// ---------------------------------------------------------------------------
// f32 -> f16 conversion pass (operands are small; lives in L2)
// ---------------------------------------------------------------------------
__global__ void cvt_f32_f16(const float* __restrict__ src,
                            _Float16* __restrict__ dst, int n) {
  const int i = blockIdx.x * blockDim.x + threadIdx.x;
  if (i < n) dst[i] = (_Float16)src[i];
}

// HsT[b][d][s] = memory_bank[s][b][d]  (f16) -- contiguous-K B operand for c-GEMM
__global__ void mk_hst(const float* __restrict__ MB, _Float16* __restrict__ HsT) {
  const int idx = blockIdx.x * blockDim.x + threadIdx.x;  // B*ENC*S
  if (idx >= Bc * ENCc * Sc) return;
  const int s = idx % Sc;
  const int rest = idx / Sc;
  const int d = rest % ENCc;
  const int b = rest / ENCc;
  HsT[idx] = (_Float16)MB[((size_t)s * Bc + b) * ENCc + d];
}

// ---------------------------------------------------------------------------
// enc_hid[b][wd] = concat(enc_state[0],enc_state[1])[b] . W_enc_out[wd] + bias
// ---------------------------------------------------------------------------
__global__ void enc_out_kernel(const float* __restrict__ enc_state,
                               const float* __restrict__ W,
                               const float* __restrict__ bias,
                               float* __restrict__ out) {
  const int idx = blockIdx.x * blockDim.x + threadIdx.x;   // B*WORD = 4096
  if (idx >= Bc * WORDc) return;
  const int b = idx / WORDc, wd = idx % WORDc;
  const float* __restrict__ w  = W + (size_t)wd * (2 * ENCc);
  const float* __restrict__ e0 = enc_state + (size_t)b * ENCc;
  const float* __restrict__ e1 = enc_state + (size_t)(Bc + b) * ENCc;
  float acc = bias[wd];
  for (int k = 0; k < ENCc; ++k) acc += e0[k] * w[k] + e1[k] * w[ENCc + k];
  out[idx] = acc;
}

// ---------------------------------------------------------------------------
// Per (b,t): align[s] = sum_d v[d]*tanh(wq[t,b,d] + uh[s,b,d]); mask; softmax.
// Writes probabilities directly as f16 rows p_h[(b*T+t), s] for the c-GEMM.
// ---------------------------------------------------------------------------
__global__ __launch_bounds__(256) void attn_kernel(
    const float* __restrict__ wq, const float* __restrict__ uh,
    const float* __restrict__ v_w, const int* __restrict__ mem_len,
    _Float16* __restrict__ p_h) {
  __shared__ float sWq[ENCc];
  __shared__ float sV[ENCc];
  __shared__ float sAlign[Sc];
  __shared__ float sRed[256];

  const int bt = blockIdx.x;
  const int b = bt / Tc, t = bt % Tc;
  const int tid = threadIdx.x;
  const int lane = tid & 31, wid = tid >> 5;

  const float* __restrict__ wqrow = wq + (size_t)(t * Bc + b) * ENCc;
  for (int d = tid; d < ENCc; d += 256) { sWq[d] = wqrow[d]; sV[d] = v_w[d]; }
  __syncthreads();

  const int len = mem_len[b];
  for (int s = wid; s < Sc; s += 8) {
    const float* __restrict__ uhrow = uh + (size_t)(s * Bc + b) * ENCc;
    float partial = 0.f;
    for (int d = lane; d < ENCc; d += 32)
      partial += sV[d] * tanhf(sWq[d] + uhrow[d]);
    for (int off = 16; off; off >>= 1) partial += __shfl_xor(partial, off);
    if (lane == 0) sAlign[s] = (s < len) ? partial : -INFINITY;
  }
  __syncthreads();

  float m = -INFINITY;
  for (int s = tid; s < Sc; s += 256) m = fmaxf(m, sAlign[s]);
  sRed[tid] = m; __syncthreads();
  for (int off = 128; off; off >>= 1) {
    if (tid < off) sRed[tid] = fmaxf(sRed[tid], sRed[tid + off]);
    __syncthreads();
  }
  const float mx = sRed[0];
  __syncthreads();

  float sum = 0.f;
  for (int s = tid; s < Sc; s += 256) {
    const float e = __expf(sAlign[s] - mx);
    sAlign[s] = e; sum += e;
  }
  sRed[tid] = sum; __syncthreads();
  for (int off = 128; off; off >>= 1) {
    if (tid < off) sRed[tid] += sRed[tid + off];
    __syncthreads();
  }
  const float inv = 1.f / sRed[0];
  _Float16* __restrict__ prow = p_h + (size_t)bt * Sc;
  for (int s = tid; s < Sc; s += 256) prow[s] = (_Float16)(sAlign[s] * inv);
}

// ---------------------------------------------------------------------------
// scores[b,t] = sigmoid( w_vrank . tanh(enc_hid[b]+cont[b,t]+word[t,b]) + b_vrank )
// ---------------------------------------------------------------------------
__global__ __launch_bounds__(128) void combine_kernel(
    const float* __restrict__ enc_hid, const float* __restrict__ cont,
    const float* __restrict__ word_h, const float* __restrict__ w_vrank,
    const float* __restrict__ b_vrank, float* __restrict__ out) {
  const int wavesPerBlock = blockDim.x >> 5;
  const int w = blockIdx.x * wavesPerBlock + (threadIdx.x >> 5);
  if (w >= Bc * Tc) return;
  const int b = w / Tc, t = w % Tc;
  const int lane = threadIdx.x & 31;
  const float* __restrict__ eh = enc_hid + (size_t)b * WORDc;
  const float* __restrict__ ch = cont + (size_t)w * WORDc;            // row b*T+t
  const float* __restrict__ wh = word_h + (size_t)(t * Bc + b) * WORDc;
  float acc = 0.f;
  for (int d = lane; d < WORDc; d += 32)
    acc += w_vrank[d] * tanhf(eh[d] + ch[d] + wh[d]);
  for (int off = 16; off; off >>= 1) acc += __shfl_xor(acc, off);
  if (lane == 0) out[w] = 1.f / (1.f + __expf(-(acc + b_vrank[0])));
}

// ---------------------------------------------------------------------------
extern "C" void kernel_launch(void* const* d_in, const int* in_sizes, int n_in,
                              void* d_out, int out_size, void* d_ws, size_t ws_size,
                              hipStream_t stream) {
  const float* enc_state   = (const float*)d_in[0];   // [2,B,ENC]
  const float* memory_bank = (const float*)d_in[1];   // [S,B,ENC]
  const float* tgt         = (const float*)d_in[2];   // [T,B,WORD]
  const int*   mem_len     = (const int*)d_in[3];     // [B]
  /* d_in[4] tgt_length: unused by the reference math */
  const float* Wq        = (const float*)d_in[5];     // [ENC,WORD]
  const float* bq        = (const float*)d_in[6];
  const float* Wc        = (const float*)d_in[7];     // [ENC,ENC]
  const float* v_w       = (const float*)d_in[8];
  const float* W_enc_out = (const float*)d_in[9];     // [WORD,2*ENC]
  const float* b_enc_out = (const float*)d_in[10];
  const float* W_enc_ctx = (const float*)d_in[11];    // [WORD,ENC]
  const float* b_enc_ctx = (const float*)d_in[12];
  const float* W_cw      = (const float*)d_in[13];    // [WORD,WORD]
  const float* b_cw      = (const float*)d_in[14];
  const float* w_vrank   = (const float*)d_in[15];
  const float* b_vrank   = (const float*)d_in[16];
  float* out = (float*)d_out;                         // [B,T,1] f32

  // ---- workspace carve-up --------------------------------------------------
  // f32 region (~12.6 MB)
  float* fws     = (float*)d_ws;
  float* enc_hid = fws;                                  // 4096
  float* wq      = enc_hid + Bc * WORDc;                 // 262144 rows (t*B+b)
  float* word_h  = wq + (size_t)Tc * Bc * ENCc;          // 262144 rows (t*B+b)
  float* uh      = word_h + (size_t)Tc * Bc * WORDc;     // 2097152 rows (s*B+b)
  float* cont    = uh + (size_t)Sc * Bc * ENCc;          // 262144 rows (b*T+t)
  float* cmat    = cont + (size_t)Bc * Tc * WORDc;       // 262144 rows (b*T+t)
  // f16 region (~11.5 MB)
  _Float16* hws    = (_Float16*)(cmat + (size_t)Bc * Tc * ENCc);
  _Float16* tgt_h  = hws;                                // 262144
  _Float16* Wq_h   = tgt_h + (size_t)Tc * Bc * WORDc;    // 262144
  _Float16* Wc_h   = Wq_h + (size_t)ENCc * WORDc;        // 262144
  _Float16* Wcw_h  = Wc_h + (size_t)ENCc * ENCc;         // 262144
  _Float16* Wctx_h = Wcw_h + (size_t)WORDc * WORDc;      // 262144
  _Float16* mb_h   = Wctx_h + (size_t)WORDc * ENCc;      // 2097152 rows (s*B+b)
  _Float16* HsT    = mb_h + (size_t)Sc * Bc * ENCc;      // 2097152 [b][d][s]
  _Float16* p_h    = HsT + (size_t)Bc * ENCc * Sc;       // 262144 rows (b*T+t)
  _Float16* c_h    = p_h + (size_t)Bc * Tc * Sc;         // 262144 rows (b*T+t)

  // ---- 0) f16 operand staging ---------------------------------------------
  cvt_f32_f16<<<1024, 256, 0, stream>>>(tgt, tgt_h, Tc * Bc * WORDc);
  cvt_f32_f16<<<1024, 256, 0, stream>>>(Wq, Wq_h, ENCc * WORDc);
  cvt_f32_f16<<<1024, 256, 0, stream>>>(Wc, Wc_h, ENCc * ENCc);
  cvt_f32_f16<<<1024, 256, 0, stream>>>(W_cw, Wcw_h, WORDc * WORDc);
  cvt_f32_f16<<<1024, 256, 0, stream>>>(W_enc_ctx, Wctx_h, WORDc * ENCc);
  cvt_f32_f16<<<8192, 256, 0, stream>>>(memory_bank, mb_h, Sc * Bc * ENCc);
  mk_hst<<<8192, 256, 0, stream>>>(memory_bank, HsT);

  // ---- 1) enc_hid ----------------------------------------------------------
  enc_out_kernel<<<16, 256, 0, stream>>>(enc_state, W_enc_out, b_enc_out, enc_hid);

  // ---- 2-4) WMMA GEMMs: wq, word_hid, uh ----------------------------------
  // M=512 strips: (512/16)*(512/64)=256 waves -> 64 blocks of 128 threads
  gemm_f16<<<64, 128, 0, stream>>>(tgt_h, Wq_h, bq, wq, Tc * Bc, ENCc, WORDc, 0, 0, 0);
  gemm_f16<<<64, 128, 0, stream>>>(tgt_h, Wcw_h, b_cw, word_h, Tc * Bc, WORDc, WORDc, 0, 0, 0);
  // M=4096: 2048 waves -> 512 blocks
  gemm_f16<<<512, 128, 0, stream>>>(mb_h, Wc_h, nullptr, uh, Sc * Bc, ENCc, ENCc, 0, 0, 0);

  // ---- 5) additive attention score + softmax -> p_h (f16) -----------------
  attn_kernel<<<Bc * Tc, 256, 0, stream>>>(wq, uh, v_w, mem_len, p_h);

  // ---- 6) c[b] = P[b] @ HsT[b]^T  (batched WMMA, 64x512x512 per b) --------
  // tiles per b: (64/16)*(512/64)=32 waves -> 8 blocks x, 8 batches y
  gemm_f16<<<dim3(8, Bc), 128, 0, stream>>>(p_h, HsT, nullptr, cmat,
                                            Tc, ENCc, Sc,
                                            (long)Tc * Sc, (long)ENCc * Sc, (long)Tc * ENCc);
  cvt_f32_f16<<<1024, 256, 0, stream>>>(cmat, c_h, Bc * Tc * ENCc);

  // ---- 7) cont_hid = c @ W_enc_ctx^T + b ----------------------------------
  gemm_f16<<<64, 128, 0, stream>>>(c_h, Wctx_h, b_enc_ctx, cont, Bc * Tc, WORDc, ENCc, 0, 0, 0);

  // ---- 8) final tanh + v-rank dot + sigmoid -------------------------------
  combine_kernel<<<128, 128, 0, stream>>>(enc_hid, cont, word_h, w_vrank, b_vrank, out);
}
